// ICN_72636486910666
// MI455X (gfx1250) — matrix operand order repaired and checked by reading, not compiled, not run on hardware
//
#include <hip/hip_runtime.h>
#include <hip/hip_bf16.h>

typedef __attribute__((ext_vector_type(16))) __bf16 bf16x16;
typedef __attribute__((ext_vector_type(8)))  __bf16 bf16x8;
typedef __attribute__((ext_vector_type(8)))  float  f32x8;

__device__ __forceinline__ f32x8 wmma_bf16(bf16x16 a, bf16x16 b, f32x8 c) {
    // 8 args: (neg_a, A, neg_b, B, c_mod, C, reuse_a, reuse_b)
    return __builtin_amdgcn_wmma_f32_16x16x32_bf16(false, a, false, b, (short)0, c, false, false);
}

// ---------------------------------------------------------------------------
// Prep kernels
// ---------------------------------------------------------------------------

// Fold the channel gathers into conv1 weights: Weff[m][o][j][k] += W1[m][o][i][c][k]
// for j = g_c(i). W1 flat layout [12][256][256][4][5] matches the id decomposition.
__global__ void scatter_w1_kernel(const float* __restrict__ W1,
                                  const int* __restrict__ poi,
                                  const int* __restrict__ demo,
                                  const int* __restrict__ tran,
                                  float* __restrict__ Wtmp) {
    size_t id = (size_t)blockIdx.x * 256 + threadIdx.x;
    if (id >= (size_t)12 * 256 * 256 * 20) return;
    int k = (int)(id % 5);
    int c = (int)((id / 5) % 4);
    int i = (int)((id / 20) % 256);
    int o = (int)((id / (20 * 256)) % 256);
    int m = (int)(id / (20 * 256 * 256));
    int j = (c == 0) ? i : (c == 1 ? poi[i] : (c == 2 ? demo[i] : tran[i]));
    atomicAdd(&Wtmp[(((size_t)m * 256 + o) * 256 + j) * 5 + k], W1[id]);
}

// Pack Weff into WMMA A-fragment order (contiguous K, matching B reads):
// element e of lane (g,m16): K = g*16+e
__global__ void pack_a1_kernel(const float* __restrict__ Wtmp, __bf16* __restrict__ A1p) {
    size_t id = (size_t)blockIdx.x * 256 + threadIdx.x;
    if (id >= (size_t)12 * 5 * 8 * 16 * 512) return;
    int e    = (int)(id % 16);
    int lane = (int)((id / 16) % 32);
    int ot   = (int)((id / 512) % 16);
    int jt   = (int)((id / 8192) % 8);
    int k    = (int)((id / 65536) % 5);
    int m    = (int)(id / 327680);
    int g = lane >> 4;
    int o = ot * 16 + (lane & 15);
    int kk = g * 16 + e;
    int j = jt * 32 + kk;
    A1p[id] = (__bf16)Wtmp[(((size_t)m * 256 + o) * 256 + j) * 5 + k];
}

// Conv2 weights need no gather: pack W2 [12][256][256][1][3] directly.
__global__ void pack_a2_kernel(const float* __restrict__ W2, __bf16* __restrict__ A2p) {
    size_t id = (size_t)blockIdx.x * 256 + threadIdx.x;
    if (id >= (size_t)12 * 3 * 8 * 16 * 512) return;
    int e    = (int)(id % 16);
    int lane = (int)((id / 16) % 32);
    int ot   = (int)((id / 512) % 16);
    int jt   = (int)((id / 8192) % 8);
    int k    = (int)((id / 65536) % 3);
    int m    = (int)(id / 196608);
    int g = lane >> 4;
    int o = ot * 16 + (lane & 15);
    int kk = g * 16 + e;
    int j = jt * 32 + kk;
    A2p[id] = (__bf16)W2[(((size_t)m * 256 + o) * 256 + j) * 3 + k];
}

// x [32][2048][256] f32 -> XE/XO [32][1024][256] bf16 (even/odd time split)
__global__ void split_x_kernel(const float* __restrict__ x,
                               __bf16* __restrict__ XE, __bf16* __restrict__ XO) {
    size_t id = (size_t)blockIdx.x * 256 + threadIdx.x;
    if (id >= (size_t)32 * 1024 * 256) return;
    int j = (int)(id % 256);
    int t = (int)((id / 256) % 1024);
    int b = (int)(id / (256 * 1024));
    size_t base = (((size_t)b * 2048 + 2 * t) * 256) + j;
    XE[id] = (__bf16)x[base];
    XO[id] = (__bf16)x[base + 256];
}

// ---------------------------------------------------------------------------
// Conv1 GEMM: h[b,o,t'] = lrelu(b1[o] + sum_k sum_j Weff[o,j,k]*X[b,j,clamp(t'+k-3)])
// Block: 256 thr = 8 waves, computes 256(o) x 64(t') tile. LDS stages 68 time rows.
// ---------------------------------------------------------------------------
__global__ __launch_bounds__(256) void conv1_kernel(
        const __bf16* __restrict__ Xin,      // [B][T2][256]
        const __bf16* __restrict__ Ap,       // packed [5][8][16][512]
        const float* __restrict__ bias,      // [256]
        __bf16* __restrict__ Hout,           // [B][T2pad][256]
        int T2, int T2pad) {
    __shared__ __attribute__((aligned(16))) __bf16 lds[68 * 256];
    const int tid = threadIdx.x;
    const int b  = blockIdx.y;
    const int t0 = blockIdx.x * 64;

    const size_t xbase = (size_t)b * T2 * 256;
    for (int chunk = tid; chunk < 68 * 32; chunk += 256) {
        int r  = chunk >> 5;
        int jc = (chunk & 31) << 3;
        int t = t0 - 3 + r;
        t = t < 0 ? 0 : (t >= T2 ? T2 - 1 : t);
        *(uint4*)(&lds[r * 256 + jc]) = *(const uint4*)(&Xin[xbase + (size_t)t * 256 + jc]);
    }
    __syncthreads();

    const int w = tid >> 5, lane = tid & 31;
    const int g = lane >> 4, n = lane & 15;
    f32x8 acc[2][4] = {};

    for (int k = 0; k < 5; ++k) {
        for (int jt = 0; jt < 8; ++jt) {
            const __bf16* ab = Ap + ((size_t)(k * 8 + jt) * 16) * 512 + (size_t)lane * 16;
            bf16x16 a0 = *(const bf16x16*)(ab + (size_t)w * 512);
            bf16x16 a1 = *(const bf16x16*)(ab + (size_t)(w + 8) * 512);
            const int ldsj = jt * 32 + g * 16;
#pragma unroll
            for (int tt = 0; tt < 4; ++tt) {
                bf16x16 bf = *(const bf16x16*)(&lds[(tt * 16 + n + k) * 256 + ldsj]);
                acc[0][tt] = wmma_bf16(a0, bf, acc[0][tt]);
                acc[1][tt] = wmma_bf16(a1, bf, acc[1][tt]);
            }
        }
    }

#pragma unroll
    for (int ai = 0; ai < 2; ++ai) {
        const int obase = (w + ai * 8) * 16 + g * 8;
#pragma unroll
        for (int tt = 0; tt < 4; ++tt) {
            const int tp = t0 + tt * 16 + n;
            bf16x8 outv;
#pragma unroll
            for (int v = 0; v < 8; ++v) {
                float h = acc[ai][tt][v] + bias[obase + v];
                h = h > 0.0f ? h : 0.01f * h;
                outv[v] = (__bf16)h;
            }
            *(bf16x8*)(&Hout[((size_t)b * T2pad + tp) * 256 + obase]) = outv;
        }
    }
}

// ---------------------------------------------------------------------------
// Conv2 GEMM + fused interactor epilogue.
// mode 0: out = other * exp(tanh(acc+b));  mode 1: other + tanh;  mode 2: other - tanh
// out_f != null  -> final fp32 write to d_out at time 4*t+phase
// out_b1 != null -> even/odd time split into out_b0/out_b1 (bf16)
// else           -> plain bf16 write to out_b0
// ---------------------------------------------------------------------------
__global__ __launch_bounds__(256) void conv2_kernel(
        const __bf16* __restrict__ Hin,      // [B][T2pad][256]
        const __bf16* __restrict__ Ap,       // packed [3][8][16][512]
        const float* __restrict__ bias,
        const __bf16* __restrict__ other,    // [B][T2][256]
        __bf16* __restrict__ out_b0, __bf16* __restrict__ out_b1,
        float* __restrict__ out_f, int phase,
        int T2, int T2pad, int mode) {
    __shared__ __attribute__((aligned(16))) __bf16 lds[66 * 256];
    const int tid = threadIdx.x;
    const int b  = blockIdx.y;
    const int t0 = blockIdx.x * 64;

    const size_t hbase = (size_t)b * T2pad * 256;
    for (int chunk = tid; chunk < 66 * 32; chunk += 256) {
        int r  = chunk >> 5;
        int jc = (chunk & 31) << 3;
        *(uint4*)(&lds[r * 256 + jc]) = *(const uint4*)(&Hin[hbase + (size_t)(t0 + r) * 256 + jc]);
    }
    __syncthreads();

    const int w = tid >> 5, lane = tid & 31;
    const int g = lane >> 4, n = lane & 15;
    f32x8 acc[2][4] = {};

    for (int k = 0; k < 3; ++k) {
        for (int jt = 0; jt < 8; ++jt) {
            const __bf16* ab = Ap + ((size_t)(k * 8 + jt) * 16) * 512 + (size_t)lane * 16;
            bf16x16 a0 = *(const bf16x16*)(ab + (size_t)w * 512);
            bf16x16 a1 = *(const bf16x16*)(ab + (size_t)(w + 8) * 512);
            const int ldsj = jt * 32 + g * 16;
#pragma unroll
            for (int tt = 0; tt < 4; ++tt) {
                bf16x16 bf = *(const bf16x16*)(&lds[(tt * 16 + n + k) * 256 + ldsj]);
                acc[0][tt] = wmma_bf16(a0, bf, acc[0][tt]);
                acc[1][tt] = wmma_bf16(a1, bf, acc[1][tt]);
            }
        }
    }

    const int T2h = T2 >> 1;
#pragma unroll
    for (int ai = 0; ai < 2; ++ai) {
        const int obase = (w + ai * 8) * 16 + g * 8;
#pragma unroll
        for (int tt = 0; tt < 4; ++tt) {
            const int t = t0 + tt * 16 + n;
            bf16x8 othv = *(const bf16x8*)(&other[((size_t)b * T2 + t) * 256 + obase]);
            float res[8];
#pragma unroll
            for (int v = 0; v < 8; ++v) {
                float tv  = tanhf(acc[ai][tt][v] + bias[obase + v]);
                float oth = (float)othv[v];
                res[v] = (mode == 0) ? oth * __expf(tv)
                       : (mode == 1) ? oth + tv
                                     : oth - tv;
            }
            if (out_f) {
                float* p = &out_f[(((size_t)b * 2048) + (size_t)(4 * t + phase)) * 256 + obase];
#pragma unroll
                for (int v = 0; v < 8; ++v) p[v] = res[v];
            } else if (out_b1) {
                __bf16* p = ((t & 1) ? out_b1 : out_b0)
                            + ((size_t)b * T2h + (t >> 1)) * 256 + obase;
#pragma unroll
                for (int v = 0; v < 8; ++v) p[v] = (__bf16)res[v];
            } else {
                __bf16* p = &out_b0[((size_t)b * T2 + t) * 256 + obase];
#pragma unroll
                for (int v = 0; v < 8; ++v) p[v] = (__bf16)res[v];
            }
        }
    }
}

// ---------------------------------------------------------------------------

extern "C" void kernel_launch(void* const* d_in, const int* in_sizes, int n_in,
                              void* d_out, int out_size, void* d_ws, size_t ws_size,
                              hipStream_t stream) {
    (void)in_sizes; (void)n_in; (void)out_size; (void)ws_size;
    const float* x    = (const float*)d_in[0];
    const float* W1   = (const float*)d_in[1];
    const float* b1   = (const float*)d_in[2];
    const float* W2   = (const float*)d_in[3];
    const float* b2   = (const float*)d_in[4];
    const int*   poi  = (const int*)d_in[5];
    const int*   demo = (const int*)d_in[6];
    const int*   tran = (const int*)d_in[7];
    float* out = (float*)d_out;

    char* wsp = (char*)d_ws;
    auto alloc = [&](size_t bytes) {
        char* p = wsp;
        wsp += (bytes + 255) & ~(size_t)255;
        return p;
    };
    __bf16* A1p  = (__bf16*)alloc((size_t)12 * 327680 * 2);       // packed conv1 weights
    __bf16* A2p  = (__bf16*)alloc((size_t)12 * 196608 * 2);       // packed conv2 weights
    float*  Wtmp = (float*) alloc((size_t)12 * 256 * 256 * 5 * 4);// gather-folded fp32 temp
    __bf16* XE   = (__bf16*)alloc((size_t)32 * 1024 * 256 * 2);
    __bf16* XO   = (__bf16*)alloc((size_t)32 * 1024 * 256 * 2);
    __bf16* Db   = (__bf16*)alloc((size_t)32 * 1024 * 256 * 2);   // d buffer (reused per level)
    __bf16* Cb   = (__bf16*)alloc((size_t)32 * 1024 * 256 * 2);   // c buffer
    __bf16* Hb   = (__bf16*)alloc((size_t)32 * 1088 * 256 * 2);   // conv1->conv2 staging
    __bf16* SEE  = (__bf16*)alloc((size_t)32 * 512 * 256 * 2);    // even(e0)
    __bf16* SEO  = (__bf16*)alloc((size_t)32 * 512 * 256 * 2);    // odd(e0)
    __bf16* SOE  = (__bf16*)alloc((size_t)32 * 512 * 256 * 2);    // even(o0)
    __bf16* SOO  = (__bf16*)alloc((size_t)32 * 512 * 256 * 2);    // odd(o0)

    // Weight preparation (fold channel gathers, pack to WMMA fragment layout)
    hipMemsetAsync(Wtmp, 0, (size_t)12 * 256 * 256 * 5 * 4, stream);
    scatter_w1_kernel<<<(12u * 256 * 256 * 20) / 256, 256, 0, stream>>>(W1, poi, demo, tran, Wtmp);
    pack_a1_kernel<<<(12u * 327680) / 256, 256, 0, stream>>>(Wtmp, A1p);
    pack_a2_kernel<<<(12u * 196608) / 256, 256, 0, stream>>>(W2, A2p);
    split_x_kernel<<<(32u * 1024 * 256) / 256, 256, 0, stream>>>(x, XE, XO);

    auto blockf = [&](int m, const __bf16* Xin, const __bf16* oth, int mode,
                      __bf16* ob0, __bf16* ob1, float* of, int phase, int T2) {
        int tg1   = (T2 + 2 + 63) / 64;   // conv1 time-groups (covers T2+2 outputs)
        int T2pad = tg1 * 64;
        int tg2   = T2 / 64;              // conv2 time-groups (exact)
        conv1_kernel<<<dim3(tg1, 32), 256, 0, stream>>>(
            Xin, A1p + (size_t)m * 327680, b1 + m * 256, Hb, T2, T2pad);
        conv2_kernel<<<dim3(tg2, 32), 256, 0, stream>>>(
            Hb, A2p + (size_t)m * 196608, b2 + m * 256, oth, ob0, ob1, of, phase, T2, T2pad, mode);
    };

    // Level 0 (node 0, T2=1024): modules phi=0, psi=1, P=2, U=3
    blockf(0, XE, XO, 0, Db,  nullptr, nullptr, 0, 1024);  // d = xo*exp(phi(xe))
    blockf(1, XO, XE, 0, Cb,  nullptr, nullptr, 0, 1024);  // c = xe*exp(psi(xo))
    blockf(3, Db, Cb, 1, SEE, SEO,     nullptr, 0, 1024);  // e0 = c + U(d), split even/odd
    blockf(2, Cb, Db, 2, SOE, SOO,     nullptr, 0, 1024);  // o0 = d - P(c), split even/odd

    // Level 1 (node 1, T2=512) on e0 -> ee (phase 0), eo (phase 2)
    blockf(4, SEE, SEO, 0, Db, nullptr, nullptr, 0, 512);
    blockf(5, SEO, SEE, 0, Cb, nullptr, nullptr, 0, 512);
    blockf(7, Db,  Cb,  1, nullptr, nullptr, out, 0, 512); // ee
    blockf(6, Cb,  Db,  2, nullptr, nullptr, out, 2, 512); // eo

    // Level 2 (node 2, T2=512) on o0 -> oe (phase 1), oo (phase 3)
    blockf(8,  SOE, SOO, 0, Db, nullptr, nullptr, 0, 512);
    blockf(9,  SOO, SOE, 0, Cb, nullptr, nullptr, 0, 512);
    blockf(11, Db,  Cb,  1, nullptr, nullptr, out, 1, 512); // oe
    blockf(10, Cb,  Db,  2, nullptr, nullptr, out, 3, 512); // oo
}
